// TransformerBlock_30236569764363
// MI455X (gfx1250) — compile-verified, matrix operand order checked by reading, not run on hardware
//
#include <hip/hip_runtime.h>
#include <math.h>

#define DME 1024
#define NCTX 2048
#define NHEAD 16
#define HDIM 64

typedef unsigned short u16;
typedef unsigned int u32;
typedef __attribute__((ext_vector_type(16))) __bf16 v16bf;
typedef __attribute__((ext_vector_type(8)))  float  v8f;
typedef __attribute__((ext_vector_type(4)))  u32    u32x4;

union BF16x16 { u32x4 u4[2]; v16bf v; };

__device__ __forceinline__ u16 f2bf(float f) {
  u32 u = __float_as_uint(f);
  u32 r = u + 0x7FFFu + ((u >> 16) & 1u);  // round-to-nearest-even
  return (u16)(r >> 16);
}

__device__ __forceinline__ v8f wmma_bf16(const BF16x16& a, const BF16x16& b, v8f c) {
  // v_wmma_f32_16x16x32_bf16 : D = A(16x32) * B(32x16) + C(16x16 f32)
  return __builtin_amdgcn_wmma_f32_16x16x32_bf16(false, a.v, false, b.v,
                                                 (short)0, c, false, false);
}

__device__ __forceinline__ void wave_lds_fence() {
  // same-wave LDS RAW fence: compiler barrier + explicit dscnt drain
  __builtin_amdgcn_wave_barrier();
  asm volatile("s_wait_dscnt 0" ::: "memory");
  __builtin_amdgcn_wave_barrier();
}

__device__ __forceinline__ float gelu_c(float x) {
  const float c = 0.79788456080286535588f;  // sqrt(2/pi)
  float t = tanhf(c * (x + 0.04715f * x * x * x));  // module's 0.04715, not 0.044715
  return 0.5f * x * (1.0f + t);
}

// ---------------- LayerNorm (f32 in -> bf16 out), one block per row ----------------
__global__ void __launch_bounds__(256)
ln_bf16_kernel(const float* __restrict__ X, const float* __restrict__ gam,
               const float* __restrict__ bet, u16* __restrict__ Y) {
  const int row  = blockIdx.x;
  const int tid  = threadIdx.x;
  const int lane = tid & 31, wave = tid >> 5;
  const float* xr = X + (size_t)row * DME;
  const int c = tid * 4;
  float4 x = *(const float4*)(xr + c);
  float s  = x.x + x.y + x.z + x.w;
  float ss = x.x * x.x + x.y * x.y + x.z * x.z + x.w * x.w;
#pragma unroll
  for (int off = 1; off < 32; off <<= 1) {
    s  += __shfl_xor(s,  off, 32);
    ss += __shfl_xor(ss, off, 32);
  }
  __shared__ float red_s[8], red_ss[8];
  if (lane == 0) { red_s[wave] = s; red_ss[wave] = ss; }
  __syncthreads();
  s = 0.f; ss = 0.f;
#pragma unroll
  for (int w = 0; w < 8; ++w) { s += red_s[w]; ss += red_ss[w]; }
  const float mean = s * (1.0f / DME);
  const float var  = ss * (1.0f / DME) - mean * mean;
  const float rstd = rsqrtf(var + 1e-5f);
  float4 g = *(const float4*)(gam + c);
  float4 b = *(const float4*)(bet + c);
  float y0 = g.x * ((x.x - mean) * rstd) + b.x;
  float y1 = g.y * ((x.y - mean) * rstd) + b.y;
  float y2 = g.z * ((x.z - mean) * rstd) + b.z;
  float y3 = g.w * ((x.w - mean) * rstd) + b.w;
  uint2 o;
  o.x = (u32)f2bf(y0) | ((u32)f2bf(y1) << 16);
  o.y = (u32)f2bf(y2) | ((u32)f2bf(y3) << 16);
  *(uint2*)(Y + (size_t)row * DME + c) = o;
}

// ------------- Weight transpose + f32->bf16 convert: WT[n][k] = W[k][n] -------------
__global__ void __launch_bounds__(256)
transpose_bf16_kernel(const float* __restrict__ W, u16* __restrict__ WT, int R, int C) {
  __shared__ float tile[32][33];
  const int bx = blockIdx.x * 32;  // column of W
  const int by = blockIdx.y * 32;  // row of W
  const int tx = threadIdx.x, ty = threadIdx.y;
#pragma unroll
  for (int j = 0; j < 4; ++j)
    tile[ty + 8 * j][tx] = W[(size_t)(by + ty + 8 * j) * C + bx + tx];
  __syncthreads();
#pragma unroll
  for (int j = 0; j < 4; ++j)
    WT[(size_t)(bx + ty + 8 * j) * R + by + tx] = f2bf(tile[tx][ty + 8 * j]);
}

// ---------------- WMMA GEMM: C[M,N] = A_bf16[M,K] @ (BT_bf16[N,K])^T ----------------
// One wave owns a 32(M) x 64(N) tile: 2 M-subtiles x 4 N-subtiles = 8 accumulators.
// Per K=32 step: 4 A-loads + 8 B-loads (16B each) feed 8 WMMAs (~22 FLOP/byte).
// EPI 0: store bf16          (QKV projections)
// EPI 1: store bf16 gelu(acc + bias)            (FFN up-proj)
// EPI 2: store f32  acc + bias + residual[M,N]  (attn out-proj / FFN down-proj)
template <int EPI>
__global__ void __launch_bounds__(256)
gemm_bf16_kernel(const u16* __restrict__ A, const u16* __restrict__ BT,
                 const float* __restrict__ bias, const float* __restrict__ res,
                 void* __restrict__ out, int M, int N, int K) {
  const int lane = threadIdx.x & 31, wave = threadIdx.x >> 5;
  const int rl = lane & 15, hi = lane >> 4, co = hi * 8;
  const int m0 = blockIdx.y * 32;
  const int n0 = blockIdx.x * 512 + wave * 64;
  const u16* arow0 = A + (size_t)(m0 + rl) * K;
  const u16* arow1 = A + (size_t)(m0 + 16 + rl) * K;
  const u16* brow0 = BT + (size_t)(n0 + rl) * K;
  v8f acc[2][4] = {};
  for (int k = 0; k < K; k += 32) {
    BF16x16 a0, a1;
    a0.u4[0] = *(const u32x4*)(arow0 + k + co);       // K = co..co+7
    a0.u4[1] = *(const u32x4*)(arow0 + k + co + 16);  // K = co+16..co+23
    a1.u4[0] = *(const u32x4*)(arow1 + k + co);
    a1.u4[1] = *(const u32x4*)(arow1 + k + co + 16);
#pragma unroll
    for (int t = 0; t < 4; ++t) {                     // B reused across both M-subtiles
      const u16* brow = brow0 + (size_t)(16 * t) * K;
      BF16x16 bm;
      bm.u4[0] = *(const u32x4*)(brow + k + co);
      bm.u4[1] = *(const u32x4*)(brow + k + co + 16);
      acc[0][t] = wmma_bf16(a0, bm, acc[0][t]);
      acc[1][t] = wmma_bf16(a1, bm, acc[1][t]);
    }
  }
#pragma unroll
  for (int mi = 0; mi < 2; ++mi)
#pragma unroll
    for (int t = 0; t < 4; ++t) {
      const int col = n0 + 16 * t + rl;
      const float bcol = (EPI != 0) ? bias[col] : 0.f;
#pragma unroll
      for (int i = 0; i < 8; ++i) {                   // C layout: M = i + 8*hi, N = rl
        const int rowm = m0 + 16 * mi + i + 8 * hi;
        float v = acc[mi][t][i];
        if (EPI == 1) v = gelu_c(v + bcol);
        if (EPI == 2) v = v + bcol + res[(size_t)rowm * N + col];
        if (EPI == 2) ((float*)out)[(size_t)rowm * N + col] = v;
        else          ((u16*)out)[(size_t)rowm * N + col] = f2bf(v);
      }
    }
}

// -------- Flash attention: one wave per 16-row q-tile, 32-key blocks, causal --------
__global__ void __launch_bounds__(256)
attn_kernel(const u16* __restrict__ Qb, const u16* __restrict__ Kb,
            const u16* __restrict__ Vb, u16* __restrict__ Zb) {
  __shared__ __align__(16) u16 sVT[8][HDIM * 40];  // V^T block, row stride 40 (80B, 16B-mult)
  __shared__ __align__(16) u16 sP[8][16 * 32];     // P tile row-major
  const int lane = threadIdx.x & 31, wave = threadIdx.x >> 5;
  const int rl = lane & 15, hi = lane >> 4, co = hi * 8;
  const int bh = blockIdx.x;
  const int bb = bh >> 4, h = bh & 15;
  const int q0 = blockIdx.y * 128 + wave * 16;
  const size_t rowBase = (size_t)bb * NCTX;

  // Q tile 16x64 as WMMA-A (two K=32 d-chunks)
  const u16* qrow = Qb + (rowBase + q0 + rl) * DME + h * HDIM;
  BF16x16 a0, a1;
  a0.u4[0] = *(const u32x4*)(qrow + co);
  a0.u4[1] = *(const u32x4*)(qrow + co + 16);
  a1.u4[0] = *(const u32x4*)(qrow + 32 + co);
  a1.u4[1] = *(const u32x4*)(qrow + 32 + co + 16);

  v8f acc[4] = {};
  float mrun[8], lrun[8];
#pragma unroll
  for (int i = 0; i < 8; ++i) { mrun[i] = -1e30f; lrun[i] = 0.f; }

  const int nkb = ((q0 + 15) >> 5) + 1;  // causal: keys 0 .. q0+15
  for (int kb = 0; kb < nkb; ++kb) {
    const int kbase = kb * 32;
    // stage V block (32 x 64) transposed into per-wave LDS
    {
      int kv = kbase + lane; if (kv > NCTX - 1) kv = NCTX - 1;
      const u16* vrow = Vb + (rowBase + kv) * DME + h * HDIM;
      u16* dst = &sVT[wave][0];
#pragma unroll
      for (int c2 = 0; c2 < 8; ++c2) {
        u32x4 t4 = *(const u32x4*)(vrow + c2 * 8);
#pragma unroll
        for (int j = 0; j < 4; ++j) {
          const int d = c2 * 8 + j * 2;
          const u32 w2 = t4[j];
          dst[(d    ) * 40 + lane] = (u16)(w2 & 0xFFFFu);
          dst[(d + 1) * 40 + lane] = (u16)(w2 >> 16);
        }
      }
    }
    // S = Q @ K^T for two 16-key tiles; B operand = contiguous rows of K
    v8f s01[2];
#pragma unroll
    for (int t = 0; t < 2; ++t) {
      int kr = kbase + 16 * t + rl; if (kr > NCTX - 1) kr = NCTX - 1;
      const u16* krow = Kb + (rowBase + kr) * DME + h * HDIM;
      BF16x16 b0, b1;
      b0.u4[0] = *(const u32x4*)(krow + co);
      b0.u4[1] = *(const u32x4*)(krow + co + 16);
      b1.u4[0] = *(const u32x4*)(krow + 32 + co);
      b1.u4[1] = *(const u32x4*)(krow + 32 + co + 16);
      v8f s = {};
      s = wmma_bf16(a0, b0, s);
      s = wmma_bf16(a1, b1, s);
      s01[t] = s;
    }
    // online softmax (row = i + 8*hi, col = rl within tile); 16-lane reductions
    float alpha[8];
#pragma unroll
    for (int i = 0; i < 8; ++i) {
      const int q = q0 + i + 8 * hi;
      float v0 = (kbase + rl      <= q) ? s01[0][i] * 0.125f : -1e30f;  // 1/sqrt(64)
      float v1 = (kbase + 16 + rl <= q) ? s01[1][i] * 0.125f : -1e30f;
      float m = fmaxf(v0, v1);
#pragma unroll
      for (int off = 1; off < 16; off <<= 1) m = fmaxf(m, __shfl_xor(m, off, 32));
      const float mnew = fmaxf(mrun[i], m);
      const float a  = __expf(mrun[i] - mnew);
      const float p0 = __expf(v0 - mnew);
      const float p1 = __expf(v1 - mnew);
      float rs = p0 + p1;
#pragma unroll
      for (int off = 1; off < 16; off <<= 1) rs += __shfl_xor(rs, off, 32);
      lrun[i] = lrun[i] * a + rs;
      mrun[i] = mnew;
      alpha[i] = a;
      s01[0][i] = p0; s01[1][i] = p1;
    }
#pragma unroll
    for (int t = 0; t < 4; ++t)
#pragma unroll
      for (int i = 0; i < 8; ++i) acc[t][i] *= alpha[i];
    // P (C layout) -> LDS row-major, then reload as WMMA-A layout
    {
      u16* pw = &sP[wave][0];
#pragma unroll
      for (int i = 0; i < 8; ++i) {
        const int r = i + 8 * hi;
        pw[r * 32 + rl]      = f2bf(s01[0][i]);
        pw[r * 32 + 16 + rl] = f2bf(s01[1][i]);
      }
    }
    wave_lds_fence();
    BF16x16 pa;
    {
      const u16* pw = &sP[wave][0];
      pa.u4[0] = *(const u32x4*)(pw + rl * 32 + co);
      pa.u4[1] = *(const u32x4*)(pw + rl * 32 + co + 16);
    }
    // Z += P @ V : B operand = rows of V^T from LDS
#pragma unroll
    for (int t2 = 0; t2 < 4; ++t2) {
      const u16* vt = &sVT[wave][(16 * t2 + rl) * 40];
      BF16x16 bv;
      bv.u4[0] = *(const u32x4*)(vt + co);
      bv.u4[1] = *(const u32x4*)(vt + co + 16);
      acc[t2] = wmma_bf16(pa, bv, acc[t2]);
    }
    wave_lds_fence();  // WAR: next iteration overwrites sVT/sP
  }
  // normalize and write Z (bf16, feeds the Wo GEMM)
#pragma unroll
  for (int t2 = 0; t2 < 4; ++t2)
#pragma unroll
    for (int i = 0; i < 8; ++i) {
      const int r = i + 8 * hi;
      const float z = acc[t2][i] / lrun[i];
      Zb[(rowBase + q0 + r) * DME + h * HDIM + 16 * t2 + rl] = f2bf(z);
    }
}

extern "C" void kernel_launch(void* const* d_in, const int* in_sizes, int n_in,
                              void* d_out, int out_size, void* d_ws, size_t ws_size,
                              hipStream_t stream) {
  (void)in_sizes; (void)n_in; (void)out_size; (void)ws_size;
  const float* X  = (const float*)d_in[0];
  const float* Wq = (const float*)d_in[1];
  const float* Wk = (const float*)d_in[2];
  const float* Wv = (const float*)d_in[3];
  const float* Wo = (const float*)d_in[4];
  const float* bo = (const float*)d_in[5];
  const float* W1 = (const float*)d_in[6];
  const float* b1 = (const float*)d_in[7];
  const float* W2 = (const float*)d_in[8];
  const float* b2 = (const float*)d_in[9];
  const float* g1 = (const float*)d_in[10];
  const float* s1 = (const float*)d_in[11];
  const float* g2 = (const float*)d_in[12];
  const float* s2 = (const float*)d_in[13];

  char* ws = (char*)d_ws;
  const size_t MB = 1024ull * 1024ull;
  u16*   WqT = (u16*)(ws + 0 * MB);
  u16*   WkT = (u16*)(ws + 2 * MB);
  u16*   WvT = (u16*)(ws + 4 * MB);
  u16*   WoT = (u16*)(ws + 6 * MB);
  u16*   W1T = (u16*)(ws + 8 * MB);    // 8 MB
  u16*   W2T = (u16*)(ws + 16 * MB);   // 8 MB
  u16*   Y   = (u16*)(ws + 24 * MB);   // 16 MB (reused for Y2)
  u16*   Qb  = (u16*)(ws + 40 * MB);   // 16 MB
  u16*   Kb  = (u16*)(ws + 56 * MB);   // 16 MB
  u16*   Vb  = (u16*)(ws + 72 * MB);   // 16 MB
  u16*   Zb  = (u16*)(ws + 88 * MB);   // 16 MB
  float* X1  = (float*)(ws + 104 * MB);// 32 MB   -> total 136 MB
  u16*   Hm  = (u16*)(ws + 40 * MB);   // FFN mid reuses dead Q..Z region (64 MB)

  const int Mrows = 4 * NCTX;  // 8192
  dim3 tb32(32, 8);
  transpose_bf16_kernel<<<dim3(DME / 32, DME / 32), tb32, 0, stream>>>(Wq, WqT, DME, DME);
  transpose_bf16_kernel<<<dim3(DME / 32, DME / 32), tb32, 0, stream>>>(Wk, WkT, DME, DME);
  transpose_bf16_kernel<<<dim3(DME / 32, DME / 32), tb32, 0, stream>>>(Wv, WvT, DME, DME);
  transpose_bf16_kernel<<<dim3(DME / 32, DME / 32), tb32, 0, stream>>>(Wo, WoT, DME, DME);
  transpose_bf16_kernel<<<dim3(4 * DME / 32, DME / 32), tb32, 0, stream>>>(W1, W1T, DME, 4 * DME);
  transpose_bf16_kernel<<<dim3(DME / 32, 4 * DME / 32), tb32, 0, stream>>>(W2, W2T, 4 * DME, DME);

  ln_bf16_kernel<<<Mrows, 256, 0, stream>>>(X, g1, s1, Y);

  dim3 gD(DME / 512, Mrows / 32);  // (2, 256)
  gemm_bf16_kernel<0><<<gD, 256, 0, stream>>>(Y, WqT, nullptr, nullptr, Qb, Mrows, DME, DME);
  gemm_bf16_kernel<0><<<gD, 256, 0, stream>>>(Y, WkT, nullptr, nullptr, Kb, Mrows, DME, DME);
  gemm_bf16_kernel<0><<<gD, 256, 0, stream>>>(Y, WvT, nullptr, nullptr, Vb, Mrows, DME, DME);

  attn_kernel<<<dim3(4 * NHEAD, NCTX / 128), 256, 0, stream>>>(Qb, Kb, Vb, Zb);

  gemm_bf16_kernel<2><<<gD, 256, 0, stream>>>(Zb, WoT, bo, X, X1, Mrows, DME, DME);

  ln_bf16_kernel<<<Mrows, 256, 0, stream>>>(X1, g2, s2, Y);

  dim3 gF(4 * DME / 512, Mrows / 32);  // (8, 256)
  gemm_bf16_kernel<1><<<gF, 256, 0, stream>>>(Y, W1T, b1, nullptr, Hm, Mrows, 4 * DME, DME);

  gemm_bf16_kernel<2><<<gD, 256, 0, stream>>>(Hm, W2T, b2, X1, (float*)d_out, Mrows, DME, 4 * DME);
}
